// AutoEnCode1_63050119906009
// MI455X (gfx1250) — matrix operand order repaired and checked by reading, not compiled, not run on hardware
//
#include <hip/hip_runtime.h>
#include <hip/hip_bf16.h>
#include <math.h>

typedef __attribute__((ext_vector_type(16))) __bf16        v16bf;
typedef __attribute__((ext_vector_type(8)))  float         v8f;
typedef __attribute__((ext_vector_type(8)))  unsigned int  v8u;
typedef __attribute__((ext_vector_type(4)))  unsigned int  v4u;
typedef __attribute__((ext_vector_type(8)))  int           v8i;
typedef __attribute__((ext_vector_type(4)))  int           v4i;

#define N_DIM 8192
#define H_DIM 1024
#define E_DIM 8192
#define PENALTY_F 10.0f

// LDS tile geometry: block = 16 rows (edges) x 256 cols, K-step 64.
#define KSTEP   64
#define LDK     72            // padded K stride (bf16 elements): 128B row + 16B pad
#define NCOLS   256           // cols per block (8 waves x 32)

__device__ __forceinline__ float sigmoidf_(float z) {
    return 1.0f / (1.0f + __expf(-z));
}

// ---- WMMA fragment loaders (wave32; A and B share contiguous k order k = 16*(lane/16) + s) ----
__device__ __forceinline__ v16bf load_a_frag(const __bf16* lds, int lane, int koff) {
    const int m  = lane & 15;
    const int kb = (lane & 16) ? 16 : 0;
    const __bf16* row = lds + m * LDK + koff + kb;
    v8u t;
#pragma unroll
    for (int v = 0; v < 8; ++v) {
        int k = 2 * v;
        t[v] = *(const unsigned int*)(row + k);
    }
    return __builtin_bit_cast(v16bf, t);
}

__device__ __forceinline__ v16bf load_b_frag(const __bf16* lds, int ncol0, int lane, int koff) {
    const int n  = ncol0 + (lane & 15);
    const int kb = (lane & 16) ? 16 : 0;
    const __bf16* row = lds + n * LDK + koff + kb;
    v8u t;
#pragma unroll
    for (int v = 0; v < 8; ++v)
        t[v] = *(const unsigned int*)(row + 2 * v);
    return __builtin_bit_cast(v16bf, t);
}

__device__ __forceinline__ unsigned long long pack4_bf16(float4 f) {
    union { __bf16 h[4]; unsigned long long u; } pk;
    pk.h[0] = (__bf16)f.x; pk.h[1] = (__bf16)f.y;
    pk.h[2] = (__bf16)f.z; pk.h[3] = (__bf16)f.w;
    return pk.u;
}

__device__ __forceinline__ float half_reduce(float t) {
    t += __shfl_xor(t, 1, 32);
    t += __shfl_xor(t, 2, 32);
    t += __shfl_xor(t, 4, 32);
    t += __shfl_xor(t, 8, 32);
    return t;
}

// ---- Tensor Data Mover: 2-D bf16 tile (tile_d0 x tile_d1 rows) -> LDS ----
// D# packing per CDNA5 ISA. Pads 16B after every 128B row
// so the LDS image has k-stride LDK=72 bf16. Issue from ONE wave per block.
__device__ __forceinline__ void tdm_load_2d_bf16(
    unsigned lds_addr, const __bf16* gptr,
    unsigned tensor_d0, unsigned tensor_d1, unsigned long long stride_d0,
    unsigned tile_d0, unsigned tile_d1)
{
    unsigned long long ga = (unsigned long long)(uintptr_t)gptr;
    v4u g0;
    g0[0] = 1u;                                               // count=1 (valid user D#)
    g0[1] = lds_addr;                                         // LDS byte address
    g0[2] = (unsigned)(ga & 0xffffffffu);                     // global_addr[31:0]
    g0[3] = (unsigned)((ga >> 32) & 0x01ffffffu) | (2u << 30); // addr[56:32] | type=2
    v8i g1;
    g1[0] = (int)((1u << 16)      // data_size = 1 -> 2 bytes
                | (1u << 20)      // pad_enable
                | (4u << 22)      // pad_interval: 1<<4 * 8B = 128B
                | (1u << 25));    // pad_amount: 2 DWORDs = 16B
    g1[1] = (int)((tensor_d0 & 0xffffu) << 16);                       // td0[15:0]
    g1[2] = (int)((tensor_d0 >> 16) | ((tensor_d1 & 0xffffu) << 16)); // td0[31:16], td1[15:0]
    g1[3] = (int)((tensor_d1 >> 16) | (tile_d0 << 16));               // td1[31:16], tile_d0
    g1[4] = (int)(tile_d1 & 0xffffu);                                 // tile_d1, tile_d2=0
    g1[5] = (int)(stride_d0 & 0xffffffffull);                         // dim0_stride[31:0]
    g1[6] = (int)((stride_d0 >> 32) & 0xffffull);                     // dim0_stride[47:32]
    g1[7] = 0;
    v4i g2 = {0, 0, 0, 0};
    v4i g3 = {0, 0, 0, 0};
    v8i g4 = {0, 0, 0, 0, 0, 0, 0, 0};
    __builtin_amdgcn_tensor_load_to_lds(g0, g1, g2, g3, g4, 0);
}

// ------------------------------- zero init -------------------------------
__global__ void zero_kernel(float* __restrict__ p, int n) {
    int i = blockIdx.x * 256 + threadIdx.x;
    if (i < n) p[i] = 0.0f;
}

// ----------------------- f32 -> bf16 weight convert ----------------------
__global__ __launch_bounds__(256)
void cvt_bf16_kernel(const float* __restrict__ src, __bf16* __restrict__ dst, int n4) {
    int i = blockIdx.x * 256 + threadIdx.x;
    if (i < n4) {
        float4 f = *(const float4*)(src + (size_t)i * 4);
        *(unsigned long long*)(dst + (size_t)i * 4) = pack4_bf16(f);
    }
}

// ------------------------------- layer 1 ---------------------------------
// Z1 = sigmoid(X @ W1^T + b1) for Xi and Xj. Block: 16 edges x 256 cols, 8 waves;
// each wave: 2 N-subtiles x {i,j} = 4 accumulators, 8 WMMAs per 64-wide K-step.
// B tile (256x64 bf16 of W1b) streamed by the Tensor Data Mover.
__global__ __launch_bounds__(256)
void gemm1_kernel(const float* __restrict__ A, const __bf16* __restrict__ W1b,
                  const float* __restrict__ b1, const int* __restrict__ edges,
                  __bf16* __restrict__ H1i, __bf16* __restrict__ H1j,
                  float* __restrict__ loss1_l1)
{
    __shared__ __bf16 sAi[16][LDK];
    __shared__ __bf16 sAj[16][LDK];
    __shared__ __bf16 sB[NCOLS][LDK];
    __shared__ int sNi[16], sNj[16];

    const int tid  = threadIdx.x;
    const int lane = tid & 31;
    const int wave = tid >> 5;
    const int mtile   = blockIdx.x;
    const int colbase = blockIdx.y * NCOLS;

    if (tid < 16) {
        sNi[tid] = edges[(mtile * 16 + tid) * 2 + 0];
        sNj[tid] = edges[(mtile * 16 + tid) * 2 + 1];
    }
    __syncthreads();

    const int ar  = tid >> 4;
    const int ac4 = (tid & 15) * 4;
    const float* gAi = A + (size_t)sNi[ar] * N_DIM + ac4;
    const float* gAj = A + (size_t)sNj[ar] * N_DIM + ac4;
    unsigned long long* lAi = (unsigned long long*)&sAi[ar][ac4];
    unsigned long long* lAj = (unsigned long long*)&sAj[ar][ac4];
    const unsigned sB_addr = (unsigned)(uintptr_t)&sB[0][0];

    v8f acc_i[2] = {};
    v8f acc_j[2] = {};

    for (int kt = 0; kt < N_DIM; kt += KSTEP) {
        // B tile: one TDM per block (wave 0 issues; TENSORcnt-tracked).
        if (wave == 0)
            tdm_load_2d_bf16(sB_addr, W1b + (size_t)colbase * N_DIM + kt,
                             N_DIM, H_DIM, N_DIM, KSTEP, NCOLS);
        // A tiles: gathered f32 rows, one float4 per thread per tile -> bf16 b64.
        *lAi = pack4_bf16(*(const float4*)(gAi + kt));
        *lAj = pack4_bf16(*(const float4*)(gAj + kt));
        if (wave == 0) __builtin_amdgcn_s_wait_tensorcnt((short)0);
        __syncthreads();

#pragma unroll
        for (int kh = 0; kh < 2; ++kh) {
            const int koff = kh * 32;
            v16bf ai = load_a_frag(&sAi[0][0], lane, koff);
            v16bf aj = load_a_frag(&sAj[0][0], lane, koff);
#pragma unroll
            for (int c = 0; c < 2; ++c) {
                v16bf bf = load_b_frag(&sB[0][0], wave * 32 + c * 16, lane, koff);
                acc_i[c] = __builtin_amdgcn_wmma_f32_16x16x32_bf16(
                    false, ai, false, bf, (short)0, acc_i[c], false, false);
                acc_j[c] = __builtin_amdgcn_wmma_f32_16x16x32_bf16(
                    false, aj, false, bf, (short)0, acc_j[c], false, false);
            }
        }
        __syncthreads();
    }

    const int n  = lane & 15;
    const int mh = (lane & 16) ? 8 : 0;
#pragma unroll
    for (int c = 0; c < 2; ++c) {
        const int col = colbase + wave * 32 + c * 16 + n;
        const float bv = b1[col];
#pragma unroll
        for (int r = 0; r < 8; ++r) {
            int m    = mh + r;
            int edge = mtile * 16 + m;
            float hi = sigmoidf_(acc_i[c][r] + bv);
            float hj = sigmoidf_(acc_j[c][r] + bv);
            H1i[(size_t)edge * H_DIM + col] = (__bf16)hi;
            H1j[(size_t)edge * H_DIM + col] = (__bf16)hj;
            float d = hi - hj;
            float t = half_reduce(d * d);
            if (n == 0) atomicAdd(&loss1_l1[edge], t);
        }
    }
}

// ------------------------------- layer 2 ---------------------------------
__global__ __launch_bounds__(256)
void gemm2_kernel(const float* __restrict__ A, const __bf16* __restrict__ W2b,
                  const float* __restrict__ b2, const int* __restrict__ edges,
                  const __bf16* __restrict__ H1i, const __bf16* __restrict__ H1j,
                  float* __restrict__ loss1_l2,
                  float* __restrict__ sxi, float* __restrict__ sxj)
{
    __shared__ __bf16 sAi[16][LDK];
    __shared__ __bf16 sAj[16][LDK];
    __shared__ __bf16 sB[NCOLS][LDK];
    __shared__ int sNi[16], sNj[16];

    const int tid  = threadIdx.x;
    const int lane = tid & 31;
    const int wave = tid >> 5;
    const int mtile   = blockIdx.x;
    const int colbase = blockIdx.y * NCOLS;

    if (tid < 16) {
        sNi[tid] = edges[(mtile * 16 + tid) * 2 + 0];
        sNj[tid] = edges[(mtile * 16 + tid) * 2 + 1];
    }
    __syncthreads();

    const int ar  = tid >> 4;
    const int ac4 = (tid & 15) * 4;
    const __bf16* gHi = H1i + (size_t)(mtile * 16 + ar) * H_DIM + ac4;
    const __bf16* gHj = H1j + (size_t)(mtile * 16 + ar) * H_DIM + ac4;
    unsigned long long* lAi = (unsigned long long*)&sAi[ar][ac4];
    unsigned long long* lAj = (unsigned long long*)&sAj[ar][ac4];
    const unsigned sB_addr = (unsigned)(uintptr_t)&sB[0][0];

    v8f acc_i[2] = {};
    v8f acc_j[2] = {};

    for (int kt = 0; kt < H_DIM; kt += KSTEP) {
        if (wave == 0)
            tdm_load_2d_bf16(sB_addr, W2b + (size_t)colbase * H_DIM + kt,
                             H_DIM, N_DIM, H_DIM, KSTEP, NCOLS);
        *lAi = *(const unsigned long long*)(gHi + kt);
        *lAj = *(const unsigned long long*)(gHj + kt);
        if (wave == 0) __builtin_amdgcn_s_wait_tensorcnt((short)0);
        __syncthreads();

#pragma unroll
        for (int kh = 0; kh < 2; ++kh) {
            const int koff = kh * 32;
            v16bf ai = load_a_frag(&sAi[0][0], lane, koff);
            v16bf aj = load_a_frag(&sAj[0][0], lane, koff);
#pragma unroll
            for (int c = 0; c < 2; ++c) {
                v16bf bf = load_b_frag(&sB[0][0], wave * 32 + c * 16, lane, koff);
                acc_i[c] = __builtin_amdgcn_wmma_f32_16x16x32_bf16(
                    false, ai, false, bf, (short)0, acc_i[c], false, false);
                acc_j[c] = __builtin_amdgcn_wmma_f32_16x16x32_bf16(
                    false, aj, false, bf, (short)0, acc_j[c], false, false);
            }
        }
        __syncthreads();
    }

    const int n  = lane & 15;
    const int mh = (lane & 16) ? 8 : 0;
#pragma unroll
    for (int c = 0; c < 2; ++c) {
        const int col = colbase + wave * 32 + c * 16 + n;
        const float bv = b2[col];
#pragma unroll
        for (int r = 0; r < 8; ++r) {
            int m    = mh + r;
            int edge = mtile * 16 + m;
            float h2i = sigmoidf_(acc_i[c][r] + bv);
            float h2j = sigmoidf_(acc_j[c][r] + bv);
            float xi  = A[(size_t)sNi[m] * N_DIM + col];
            float xj  = A[(size_t)sNj[m] * N_DIM + col];
            float dh  = h2i - h2j;
            float di  = xi - h2i;
            float dj  = xj - h2j;
            float th = half_reduce(dh * dh);
            float ti = half_reduce(di * di);
            float tj = half_reduce(dj * dj);
            if (n == 0) {
                atomicAdd(&loss1_l2[edge], th);
                atomicAdd(&sxi[edge], ti);
                atomicAdd(&sxj[edge], tj);
            }
        }
    }
}

// --------------------------- regularizer norms ---------------------------
__global__ __launch_bounds__(256)
void lossr_kernel(const float* __restrict__ W1, const float* __restrict__ b1,
                  const float* __restrict__ W2, const float* __restrict__ b2,
                  float* __restrict__ lossr)
{
    __shared__ float red[256];
    const int bid = blockIdx.x;
    const float* ptr;
    int len;
    if (bid < H_DIM)               { ptr = W1 + (size_t)bid * N_DIM;            len = N_DIM; }
    else if (bid < H_DIM + N_DIM)  { ptr = W2 + (size_t)(bid - H_DIM) * H_DIM;  len = H_DIM; }
    else if (bid == H_DIM + N_DIM) { ptr = b1;                                  len = H_DIM; }
    else                           { ptr = b2;                                  len = N_DIM; }

    float s = 0.0f;
    for (int i = threadIdx.x; i < len; i += 256) { float v = ptr[i]; s += v * v; }
    red[threadIdx.x] = s;
    __syncthreads();
    for (int off = 128; off > 0; off >>= 1) {
        if (threadIdx.x < off) red[threadIdx.x] += red[threadIdx.x + off];
        __syncthreads();
    }
    if (threadIdx.x == 0) atomicAdd(lossr, sqrtf(red[0]));
}

// ------------------------------ final reduce -----------------------------
__global__ __launch_bounds__(256)
void final_kernel(const float* __restrict__ l1, const float* __restrict__ l2,
                  const float* __restrict__ sxi, const float* __restrict__ sxj,
                  const int* __restrict__ labels, const float* __restrict__ lossr,
                  float* __restrict__ out)
{
    __shared__ float red[256];
    float s = 0.0f;
    for (int e = threadIdx.x; e < E_DIM; e += 256) {
        float lab = (float)labels[e];
        float v = lab * (sqrtf(l1[e]) + sqrtf(l2[e]));
        float factor = (lab >= 1.0f) ? PENALTY_F : 1.0f;
        v += factor * (sqrtf(sxi[e]) + sqrtf(sxj[e]));
        s += v;
    }
    red[threadIdx.x] = s;
    __syncthreads();
    for (int off = 128; off > 0; off >>= 1) {
        if (threadIdx.x < off) red[threadIdx.x] += red[threadIdx.x + off];
        __syncthreads();
    }
    if (threadIdx.x == 0) out[0] = red[0] + (float)E_DIM * lossr[0];
}

// ------------------------------- launcher --------------------------------
extern "C" void kernel_launch(void* const* d_in, const int* in_sizes, int n_in,
                              void* d_out, int out_size, void* d_ws, size_t ws_size,
                              hipStream_t stream)
{
    const float* A      = (const float*)d_in[0];
    const float* W1     = (const float*)d_in[1];
    const float* b1     = (const float*)d_in[2];
    const float* W2     = (const float*)d_in[3];
    const float* b2     = (const float*)d_in[4];
    const int*   edges  = (const int*)d_in[5];
    const int*   labels = (const int*)d_in[6];
    float* out = (float*)d_out;

    char* ws = (char*)d_ws;
    __bf16* W1b = (__bf16*)ws;                                        // 16 MB
    __bf16* W2b = (__bf16*)(ws + (size_t)H_DIM * N_DIM * 2);          // 16 MB
    __bf16* H1i = (__bf16*)(ws + (size_t)H_DIM * N_DIM * 4);          // 16 MB
    __bf16* H1j = (__bf16*)(ws + (size_t)H_DIM * N_DIM * 4
                               + (size_t)E_DIM * H_DIM * 2);          // 16 MB
    float*  acc = (float*)(ws + (size_t)H_DIM * N_DIM * 4
                              + (size_t)E_DIM * H_DIM * 4);           // 4*E + 1
    float* l1    = acc;
    float* l2    = acc + E_DIM;
    float* sxi   = acc + 2 * E_DIM;
    float* sxj   = acc + 3 * E_DIM;
    float* lossr = acc + 4 * E_DIM;

    const int nacc = 4 * E_DIM + 1;
    zero_kernel<<<(nacc + 255) / 256, 256, 0, stream>>>(acc, nacc);

    const int w_n4 = (H_DIM * N_DIM) / 4;  // 2M float4 per weight matrix
    cvt_bf16_kernel<<<(w_n4 + 255) / 256, 256, 0, stream>>>(W1, W1b, w_n4);
    cvt_bf16_kernel<<<(w_n4 + 255) / 256, 256, 0, stream>>>(W2, W2b, w_n4);

    gemm1_kernel<<<dim3(E_DIM / 16, H_DIM / NCOLS), 256, 0, stream>>>(
        A, W1b, b1, edges, H1i, H1j, l1);

    gemm2_kernel<<<dim3(E_DIM / 16, N_DIM / NCOLS), 256, 0, stream>>>(
        A, W2b, b2, edges, H1i, H1j, l2, sxi, sxj);

    lossr_kernel<<<H_DIM + N_DIM + 2, 256, 0, stream>>>(W1, b1, W2, b2, lossr);

    final_kernel<<<1, 256, 0, stream>>>(l1, l2, sxi, sxj, labels, lossr, out);
}
